// BertLinear_1254130450807
// MI455X (gfx1250) — matrix operand order repaired and checked by reading, not compiled, not run on hardware
//
#include <hip/hip_runtime.h>

#define B_  8
#define S_  2048
#define D_  768
#define H_  12
#define DH_ 64
#define L_  12
#define NC_ 8            // S-chunks for kv reduction
#define LN_EPS 1e-12f
#define EPS_   1e-6f

typedef __attribute__((ext_vector_type(16))) __bf16 v16bf;
typedef __attribute__((ext_vector_type(8)))  float  v8f;

__device__ __forceinline__ float fastrcp(float x) { return __builtin_amdgcn_rcpf(x); }
__device__ __forceinline__ float siluf(float x)   { return x * fastrcp(1.0f + __expf(-x)); }

// Load a 16-element bf16 fragment as two contiguous 16B chunks (ds_load_b128 x2).
__device__ __forceinline__ v16bf load_frag(const __bf16* p0, const __bf16* p1) {
    union { v16bf v; int4 q[2]; } u;
    u.q[0] = *(const int4*)p0;
    u.q[1] = *(const int4*)p1;
    return u.v;
}

// ---------------------------------------------------------------------------
// One-time weight convert + transpose: W[k][n] fp32 -> WT[n][k] bf16.
// grid (D/32, D/32, 3*L); 256 threads; LDS 32x32 tile.
// ---------------------------------------------------------------------------
__global__ void wcvt_kernel(const float* __restrict__ Wq, const float* __restrict__ Wk,
                            const float* __restrict__ Wv, __bf16* __restrict__ WT) {
    const int z = blockIdx.z;              // z = mode*L + l
    const int l = z % L_, mode = z / L_;
    const float* src = ((mode == 0) ? Wq : (mode == 1) ? Wk : Wv) + (size_t)l * D_ * D_;
    __bf16* dst = WT + (size_t)z * D_ * D_;
    const int k0 = blockIdx.x * 32, n0 = blockIdx.y * 32;
    const int t = threadIdx.x;
    __shared__ float ts[32][33];
    #pragma unroll
    for (int i = 0; i < 4; ++i) {
        int e = t + i * 256, kk = e >> 5, nn = e & 31;
        ts[kk][nn] = src[(size_t)(k0 + kk) * D_ + n0 + nn];
    }
    __syncthreads();
    #pragma unroll
    for (int i = 0; i < 4; ++i) {
        int e = t + i * 256, nn = e >> 5, kk = e & 31;
        dst[(size_t)(n0 + nn) * D_ + k0 + kk] = (__bf16)ts[kk][nn];
    }
}

// ---------------------------------------------------------------------------
// Embedding gather + LayerNorm  (one block per (b,s) row); emits fp32 + bf16.
// ---------------------------------------------------------------------------
__global__ void embed_ln_kernel(const int* __restrict__ tok, const int* __restrict__ typ,
                                const float* __restrict__ wemb, const float* __restrict__ temb,
                                const float* __restrict__ pemb,
                                const float* __restrict__ g, const float* __restrict__ be,
                                float* __restrict__ h, __bf16* __restrict__ hbf) {
    int row = blockIdx.x;
    int t   = threadIdx.x;
    int s   = row % S_;
    int tk  = tok[row], ty = typ[row];
    __shared__ float xs[D_];
    __shared__ float red[256], red2[256];
    float sum = 0.f, sq = 0.f;
    for (int n = t; n < D_; n += 256) {
        float v = wemb[(size_t)tk * D_ + n] + temb[(size_t)ty * D_ + n] + pemb[(size_t)s * D_ + n];
        xs[n] = v; sum += v; sq += v * v;
    }
    red[t] = sum; red2[t] = sq;
    __syncthreads();
    for (int o = 128; o > 0; o >>= 1) {
        if (t < o) { red[t] += red[t + o]; red2[t] += red2[t + o]; }
        __syncthreads();
    }
    float mean = red[0] * (1.0f / D_);
    float var  = red2[0] * (1.0f / D_) - mean * mean;
    float rstd = rsqrtf(var + LN_EPS);
    for (int n = t; n < D_; n += 256) {
        float v = (xs[n] - mean) * rstd * g[n] + be[n];
        h[(size_t)row * D_ + n]   = v;
        hbf[(size_t)row * D_ + n] = (__bf16)v;
    }
}

// ---------------------------------------------------------------------------
// QKV GEMM: (16384 x 768) @ (768 x 768) + bias, fused silu / mask epilogue.
// Block tile 64x128, 256 threads = 8 waves; wave owns 1 m-tile x 4 n-tiles.
// K-step 64, double-buffered LDS, register-prefetch pipeline.
// Last iteration peeled so the steady-state body is branchless: prefetch
// loads stay pinned above the WMMA block by sched_barriers in one BB.
// ---------------------------------------------------------------------------
__global__ void __launch_bounds__(256, 1)
qkv_gemm_kernel(const __bf16* __restrict__ hbf,
                const __bf16* __restrict__ WT,
                const float* __restrict__ bq, const float* __restrict__ bk,
                const float* __restrict__ bv,
                const int*   __restrict__ tok, int layer,
                __bf16* __restrict__ qf, __bf16* __restrict__ kf,
                __bf16* __restrict__ vbuf) {
    const int mode = blockIdx.z;                    // 0=Q, 1=K, 2=V (uniform per block)
    const __bf16* W   = WT + (size_t)(mode * L_ + layer) * D_ * D_;
    const float* bias = (mode == 0) ? bq : (mode == 1) ? bk : bv;
    __bf16*      out  = (mode == 0) ? qf : (mode == 1) ? kf : vbuf;

    const int rowBase = blockIdx.x * 64;
    const int nBase   = blockIdx.y * 128;
    const int t    = threadIdx.x;
    const int lane = t & 31, wave = t >> 5;
    const int hi   = lane >> 4, lm = lane & 15;
    const int ti   = wave >> 1;                     // m-tile 0..3
    const int tjB  = (wave & 1) * 4;                // n-tiles tjB..tjB+3

    __shared__ __align__(16) __bf16 aLds[2][64][64];    // [buf][m][k]
    __shared__ __align__(16) __bf16 bT[2][128][64];     // [buf][n][k]

    // per-thread staging (K-step 64): A 64x64 bf16 = 512 int4 (2/thread),
    //                                 B 128x64 bf16 = 1024 int4 (4/thread)
    const int aM = t >> 3,          aK8 = (t & 7) * 8;          // + i*32 rows
    const int bN = t >> 3,          bK8 = (t & 7) * 8;          // + i*32 rows
    int4 aR[2], bR[4];
    v8f acc[4] = {{}, {}, {}, {}};

    const int m = ti * 16 + lm;
    const int n0 = (tjB + 0) * 16 + lm;
    const int n1 = (tjB + 1) * 16 + lm;
    const int n2 = (tjB + 2) * 16 + lm;
    const int n3 = (tjB + 3) * 16 + lm;

    // ---- prologue: tile 0 -> regs -> buffer 0 ----
    #pragma unroll
    for (int i = 0; i < 2; ++i)
        aR[i] = *(const int4*)&hbf[(size_t)(rowBase + aM + i * 32) * D_ + aK8];
    #pragma unroll
    for (int i = 0; i < 4; ++i)
        bR[i] = *(const int4*)&W[(size_t)(nBase + bN + i * 32) * D_ + bK8];
    #pragma unroll
    for (int i = 0; i < 2; ++i) *(int4*)&aLds[0][aM + i * 32][aK8] = aR[i];
    #pragma unroll
    for (int i = 0; i < 4; ++i) *(int4*)&bT[0][bN + i * 32][bK8] = bR[i];
    __syncthreads();

    int cur = 0;
    #pragma unroll 1
    for (int it = 0; it < D_ / 64 - 1; ++it) {
        const int kk = (it + 1) * 64;
        // unconditional prefetch: loads issued before the WMMA block
        #pragma unroll
        for (int i = 0; i < 2; ++i)
            aR[i] = *(const int4*)&hbf[(size_t)(rowBase + aM + i * 32) * D_ + kk + aK8];
        #pragma unroll
        for (int i = 0; i < 4; ++i)
            bR[i] = *(const int4*)&W[(size_t)(nBase + bN + i * 32) * D_ + kk + bK8];
        __builtin_amdgcn_sched_barrier(0);

        // compute: 2 K-chunks of 32, 4 n-tiles each
        #pragma unroll
        for (int c = 0; c < 2; ++c) {
            v16bf a = load_frag(&aLds[cur][m][c * 32 + hi * 16],
                                &aLds[cur][m][c * 32 + hi * 16 + 8]);
            v16bf b0 = load_frag(&bT[cur][n0][c * 32 + hi * 16], &bT[cur][n0][c * 32 + hi * 16 + 8]);
            acc[0] = __builtin_amdgcn_wmma_f32_16x16x32_bf16(false, a, false, b0, (short)0, acc[0], false, false);
            v16bf b1 = load_frag(&bT[cur][n1][c * 32 + hi * 16], &bT[cur][n1][c * 32 + hi * 16 + 8]);
            acc[1] = __builtin_amdgcn_wmma_f32_16x16x32_bf16(false, a, false, b1, (short)0, acc[1], false, false);
            v16bf b2 = load_frag(&bT[cur][n2][c * 32 + hi * 16], &bT[cur][n2][c * 32 + hi * 16 + 8]);
            acc[2] = __builtin_amdgcn_wmma_f32_16x16x32_bf16(false, a, false, b2, (short)0, acc[2], false, false);
            v16bf b3 = load_frag(&bT[cur][n3][c * 32 + hi * 16], &bT[cur][n3][c * 32 + hi * 16 + 8]);
            acc[3] = __builtin_amdgcn_wmma_f32_16x16x32_bf16(false, a, false, b3, (short)0, acc[3], false, false);
        }
        __builtin_amdgcn_sched_barrier(0);

        // unconditional store of prefetched regs into the other buffer
        const int nxt = cur ^ 1;
        #pragma unroll
        for (int i = 0; i < 2; ++i) *(int4*)&aLds[nxt][aM + i * 32][aK8] = aR[i];
        #pragma unroll
        for (int i = 0; i < 4; ++i) *(int4*)&bT[nxt][bN + i * 32][bK8] = bR[i];
        __syncthreads();
        cur = nxt;
    }

    // peeled final tile: compute only
    #pragma unroll
    for (int c = 0; c < 2; ++c) {
        v16bf a = load_frag(&aLds[cur][m][c * 32 + hi * 16],
                            &aLds[cur][m][c * 32 + hi * 16 + 8]);
        v16bf b0 = load_frag(&bT[cur][n0][c * 32 + hi * 16], &bT[cur][n0][c * 32 + hi * 16 + 8]);
        acc[0] = __builtin_amdgcn_wmma_f32_16x16x32_bf16(false, a, false, b0, (short)0, acc[0], false, false);
        v16bf b1 = load_frag(&bT[cur][n1][c * 32 + hi * 16], &bT[cur][n1][c * 32 + hi * 16 + 8]);
        acc[1] = __builtin_amdgcn_wmma_f32_16x16x32_bf16(false, a, false, b1, (short)0, acc[1], false, false);
        v16bf b2 = load_frag(&bT[cur][n2][c * 32 + hi * 16], &bT[cur][n2][c * 32 + hi * 16 + 8]);
        acc[2] = __builtin_amdgcn_wmma_f32_16x16x32_bf16(false, a, false, b2, (short)0, acc[2], false, false);
        v16bf b3 = load_frag(&bT[cur][n3][c * 32 + hi * 16], &bT[cur][n3][c * 32 + hi * 16 + 8]);
        acc[3] = __builtin_amdgcn_wmma_f32_16x16x32_bf16(false, a, false, b3, (short)0, acc[3], false, false);
    }

    // epilogue: bias + silu (+mask) + bf16 store.
    #pragma unroll
    for (int r = 0; r < 8; ++r) {
        int mg = rowBase + ti * 16 + r + 8 * hi;
        float msk = 1.0f;
        if (mode == 1) msk = (tok[mg] != 0) ? 1.0f : 0.0f;
        #pragma unroll
        for (int j = 0; j < 4; ++j) {
            int ng = nBase + (tjB + j) * 16 + lm;
            float v = acc[j][r] + bias[ng];
            if (mode <= 1) v = siluf(v);
            if (mode == 1) v *= msk;
            out[(size_t)mg * D_ + ng] = (__bf16)v;
        }
    }
}

// ---------------------------------------------------------------------------
// Partial kv[b,h] = kf^T @ v over an S-chunk of 256 (WMMA); partial ksum fused.
// Grid (NC_, B*H); 128 threads = 4 waves; wave w owns d-tile w x 4 e-tiles.
// ---------------------------------------------------------------------------
__global__ void kv_partial_kernel(const __bf16* __restrict__ kf, const __bf16* __restrict__ vbuf,
                                  float* __restrict__ kvp, float* __restrict__ ksp) {
    const int chunk = blockIdx.x;
    const int bh = blockIdx.y;
    const int b = bh / H_, hh = bh % H_;
    const int t = threadIdx.x;
    const int lane = t & 31, wave = t >> 5;
    const int hi = lane >> 4, lm = lane & 15;

    __shared__ __align__(16) __bf16 kfT[64][32];   // [d][s]  (A = kf^T)
    __shared__ __align__(16) __bf16 vT[64][32];    // [e][s]  (B = v, transposed stage)

    v8f acc[4] = {{}, {}, {}, {}};
    float ks = 0.0f;
    const size_t base = (size_t)b * S_ * D_ + (size_t)hh * DH_;
    const int sBeg = chunk * (S_ / NC_);

    for (int s0 = sBeg; s0 < sBeg + S_ / NC_; s0 += 32) {
        #pragma unroll
        for (int e = t; e < 1024; e += 128) {
            int s = e >> 5, d2 = (e & 31) * 2;
            size_t gi = base + (size_t)(s0 + s) * D_ + d2;
            __bf16 k0 = kf[gi], k1 = kf[gi + 1];
            __bf16 v0 = vbuf[gi], v1 = vbuf[gi + 1];
            kfT[d2][s] = k0; kfT[d2 + 1][s] = k1;
            vT[d2][s]  = v0; vT[d2 + 1][s]  = v1;
        }
        __syncthreads();
        if (t < 64) {
            #pragma unroll
            for (int s = 0; s < 32; ++s) ks += (float)kfT[t][s];
        }
        int m = wave * 16 + lm;
        v16bf a = load_frag(&kfT[m][hi * 16], &kfT[m][hi * 16 + 8]);
        #pragma unroll
        for (int j = 0; j < 4; ++j) {
            int n = j * 16 + lm;
            v16bf bf = load_frag(&vT[n][hi * 16], &vT[n][hi * 16 + 8]);
            acc[j] = __builtin_amdgcn_wmma_f32_16x16x32_bf16(false, a, false, bf,
                                                             (short)0, acc[j], false, false);
        }
        __syncthreads();
    }

    float* kvo = kvp + ((size_t)chunk * (B_ * H_) + bh) * DH_ * DH_;
    #pragma unroll
    for (int j = 0; j < 4; ++j) {
        #pragma unroll
        for (int r = 0; r < 8; ++r) {
            int mm = wave * 16 + r + 8 * hi;
            int n = j * 16 + lm;
            kvo[mm * DH_ + n] = acc[j][r];
        }
    }
    if (t < 64) ksp[((size_t)chunk * (B_ * H_) + bh) * DH_ + t] = ks;
}

// Reduce NC_ partial kv / ksum chunks. One block per (b,h), 256 threads.
__global__ void kv_reduce_kernel(const float* __restrict__ kvp, const float* __restrict__ ksp,
                                 float* __restrict__ kv, float* __restrict__ ksum) {
    const int bh = blockIdx.x;
    const int t  = threadIdx.x;
    for (int i = t; i < DH_ * DH_; i += 256) {
        float s = 0.f;
        #pragma unroll
        for (int c = 0; c < NC_; ++c)
            s += kvp[((size_t)c * (B_ * H_) + bh) * DH_ * DH_ + i];
        kv[(size_t)bh * DH_ * DH_ + i] = s;
    }
    for (int i = t; i < DH_; i += 256) {
        float s = 0.f;
        #pragma unroll
        for (int c = 0; c < NC_; ++c)
            s += ksp[((size_t)c * (B_ * H_) + bh) * DH_ + i];
        ksum[(size_t)bh * DH_ + i] = s;
    }
}

// ---------------------------------------------------------------------------
// att = (qf @ kv) * rcp(qf . ksum + eps); residual + LayerNorm. Block per row.
// Emits fp32 hout (+ bf16 mirror for the next layer's GEMM).
// ---------------------------------------------------------------------------
__global__ void attn_ln_kernel(const __bf16* __restrict__ qf,
                               const float* __restrict__ kv, const float* __restrict__ ksum,
                               const float* __restrict__ hin, float* __restrict__ hout,
                               __bf16* __restrict__ hbf,
                               const float* __restrict__ g, const float* __restrict__ be) {
    const int row = blockIdx.x;
    const int b   = row / S_;
    const int t   = threadIdx.x;
    __shared__ float qs[D_];
    __shared__ float xs[D_];
    __shared__ float rden[H_];
    __shared__ float red[256], red2[256];

    // stage qf row, 8 bf16 per int4 load
    for (int i = t; i < D_ / 8; i += 256) {
        union { int4 q; __bf16 b[8]; } u;
        u.q = *(const int4*)&qf[(size_t)row * D_ + i * 8];
        #pragma unroll
        for (int c = 0; c < 8; ++c) qs[i * 8 + c] = (float)u.b[c];
    }
    __syncthreads();
    if (t < H_) {
        const float* ksp = ksum + ((size_t)b * H_ + t) * DH_;
        float d = 0.f;
        #pragma unroll
        for (int i = 0; i < DH_; ++i) d += qs[t * DH_ + i] * ksp[i];
        rden[t] = fastrcp(d + EPS_);
    }
    __syncthreads();

    float sum = 0.f, sq = 0.f;
    for (int n = t; n < D_; n += 256) {
        int hh = n / DH_, e = n % DH_;
        const float* kvp = kv + ((size_t)b * H_ + hh) * DH_ * DH_ + e;
        float num = 0.f;
        #pragma unroll
        for (int d = 0; d < DH_; ++d) num += qs[hh * DH_ + d] * kvp[d * DH_];
        float v = hin[(size_t)row * D_ + n] + num * rden[hh];
        xs[n] = v; sum += v; sq += v * v;
    }
    red[t] = sum; red2[t] = sq;
    __syncthreads();
    for (int o = 128; o > 0; o >>= 1) {
        if (t < o) { red[t] += red[t + o]; red2[t] += red2[t + o]; }
        __syncthreads();
    }
    float mean = red[0] * (1.0f / D_);
    float var  = red2[0] * (1.0f / D_) - mean * mean;
    float rstd = rsqrtf(var + LN_EPS);
    for (int n = t; n < D_; n += 256) {
        float v = (xs[n] - mean) * rstd * g[n] + be[n];
        hout[(size_t)row * D_ + n] = v;
        hbf[(size_t)row * D_ + n]  = (__bf16)v;
    }
}

// ---------------------------------------------------------------------------
extern "C" void kernel_launch(void* const* d_in, const int* in_sizes, int n_in,
                              void* d_out, int out_size, void* d_ws, size_t ws_size,
                              hipStream_t stream) {
    (void)in_sizes; (void)n_in; (void)out_size; (void)ws_size;
    const int*   tok  = (const int*)d_in[0];
    const int*   typ  = (const int*)d_in[1];
    const float* wemb = (const float*)d_in[2];
    const float* temb = (const float*)d_in[3];
    const float* pemb = (const float*)d_in[4];
    const float* leg  = (const float*)d_in[5];
    const float* leb  = (const float*)d_in[6];
    const float* Wq   = (const float*)d_in[7];
    const float* bq   = (const float*)d_in[8];
    const float* Wk   = (const float*)d_in[9];
    const float* bk   = (const float*)d_in[10];
    const float* Wv   = (const float*)d_in[11];
    const float* bv   = (const float*)d_in[12];
    const float* lg   = (const float*)d_in[13];
    const float* lb   = (const float*)d_in[14];

    char* ws = (char*)d_ws;
    size_t off = 0;
    float*  h    = (float*)(ws + off);  off += (size_t)B_ * S_ * D_ * 4;
    __bf16* hbf  = (__bf16*)(ws + off); off += (size_t)B_ * S_ * D_ * 2;
    __bf16* qf   = (__bf16*)(ws + off); off += (size_t)B_ * S_ * D_ * 2;
    __bf16* kf   = (__bf16*)(ws + off); off += (size_t)B_ * S_ * D_ * 2;
    __bf16* vbuf = (__bf16*)(ws + off); off += (size_t)B_ * S_ * D_ * 2;
    __bf16* WT   = (__bf16*)(ws + off); off += (size_t)3 * L_ * D_ * D_ * 2;
    float*  kv   = (float*)(ws + off);  off += (size_t)B_ * H_ * DH_ * DH_ * 4;
    float*  ksum = (float*)(ws + off);  off += (size_t)B_ * H_ * DH_ * 4;
    float*  kvp  = (float*)(ws + off);  off += (size_t)NC_ * B_ * H_ * DH_ * DH_ * 4;
    float*  ksp  = (float*)(ws + off);  off += (size_t)NC_ * B_ * H_ * DH_ * 4;

    const int rows = B_ * S_;

    dim3 gw(D_ / 32, D_ / 32, 3 * L_);
    wcvt_kernel<<<gw, 256, 0, stream>>>(Wq, Wk, Wv, WT);
    embed_ln_kernel<<<rows, 256, 0, stream>>>(tok, typ, wemb, temb, pemb, leg, leb, h, hbf);

    for (int l = 0; l < L_; ++l) {
        const float* bql = bq + (size_t)l * D_;
        const float* bkl = bk + (size_t)l * D_;
        const float* bvl = bv + (size_t)l * D_;

        dim3 gq(rows / 64, D_ / 128, 3);
        qkv_gemm_kernel<<<gq, 256, 0, stream>>>(hbf, WT, bql, bkl, bvl, tok, l, qf, kf, vbuf);

        dim3 gkv(NC_, B_ * H_);
        kv_partial_kernel<<<gkv, 128, 0, stream>>>(kf, vbuf, kvp, ksp);
        kv_reduce_kernel<<<B_ * H_, 256, 0, stream>>>(kvp, ksp, kv, ksum);

        float* hout = (l == L_ - 1) ? (float*)d_out : h;
        attn_ln_kernel<<<rows, 256, 0, stream>>>(qf, kv, ksum, h, hout, hbf,
                                                 lg + (size_t)l * D_, lb + (size_t)l * D_);
    }
}
